// ConditionedLM_15307263443414
// MI455X (gfx1250) — matrix operand order repaired and checked by reading, not compiled, not run on hardware
//
#include <hip/hip_runtime.h>

// ---------- types ----------
typedef __attribute__((ext_vector_type(16))) __bf16        v16bf;
typedef __attribute__((ext_vector_type(8)))  float         v8f;
typedef __attribute__((ext_vector_type(4)))  unsigned int  u32x4;
typedef __attribute__((ext_vector_type(2)))  unsigned int  u32x2;
typedef __attribute__((ext_vector_type(4)))  float         f32x4;
typedef unsigned short u16;

// ---------- helpers ----------
__device__ __forceinline__ u16 f2bf(float x) {
  unsigned int u = __float_as_uint(x);
  unsigned int r = u + 0x7FFFu + ((u >> 16) & 1u);   // round-to-nearest-even
  return (u16)(r >> 16);
}
__device__ __forceinline__ float sigm(float x) { return 1.0f / (1.0f + __expf(-x)); }
__device__ __forceinline__ float tanh_(float x) {
  float e = __expf(2.0f * x);
  return 1.0f - 2.0f / (e + 1.0f);
}

union V16U { u32x4 u[2]; v16bf v; };

// D = A(16x32 bf16) * B(32x16 bf16) + C(f32), wave32 WMMA
__device__ __forceinline__ v8f wmma_bf16(u32x4 a0, u32x4 a1, u32x4 b0, u32x4 b1, v8f c) {
  V16U A, B;
  A.u[0] = a0; A.u[1] = a1;
  B.u[0] = b0; B.u[1] = b1;
  return __builtin_amdgcn_wmma_f32_16x16x32_bf16(false, A.v, false, B.v, (short)0, c, false, false);
}

// ---------- elementwise f32 -> bf16 ----------
__global__ void cvt_bf16_kernel(const float* __restrict__ in, u16* __restrict__ out, long long n) {
  long long i = (long long)blockIdx.x * blockDim.x + threadIdx.x;
  if (i < n) out[i] = f2bf(in[i]);
}

// ---------- embedding gather: ids [Bn, L] -> out[(l*Bn+b)*E + e] = emb[ids[b,l], e] (bf16) ----------
__global__ void gather_embed_kernel(const int* __restrict__ ids, int Bn, int L,
                                    const float* __restrict__ emb, u16* __restrict__ out, int E) {
  long long gid = (long long)blockIdx.x * blockDim.x + threadIdx.x;
  long long total = (long long)L * Bn * E;
  if (gid >= total) return;
  int e = (int)(gid % E);
  long long row = gid / E;
  int l = (int)(row / Bn), b = (int)(row % Bn);
  int tok = ids[b * L + l];
  out[gid] = f2bf(emb[(long long)tok * E + e]);
}

// ---------- C[M,N] = A[M,K](bf16) * W[N,K]^T(bf16) + bias[N]; grid(M/16, N/64), 128 thr ----------
__global__ __launch_bounds__(128) void gemm_bf16_kernel(
    const u16* __restrict__ A, const u16* __restrict__ W,
    const float* __restrict__ bias, float* __restrict__ C, int K, int N) {
  int tid = threadIdx.x, wave = tid >> 5, lane = tid & 31;
  int laneM = lane & 15, hi = lane >> 4;
  int m0 = blockIdx.x * 16;
  int n0 = blockIdx.y * 64 + wave * 16;
  const u16* arow = A + (size_t)(m0 + laneM) * K + hi * 16;
  const u16* brow = W + (size_t)(n0 + laneM) * K + hi * 16;
  v8f acc = {};
  for (int k0 = 0; k0 < K; k0 += 32) {
    u32x4 a0 = *(const u32x4*)(arow + k0);
    u32x4 a1 = *(const u32x4*)(arow + k0 + 8);
    u32x4 b0 = *(const u32x4*)(brow + k0);
    u32x4 b1 = *(const u32x4*)(brow + k0 + 8);
    acc = wmma_bf16(a0, a1, b0, b1, acc);
  }
  float bv = bias ? bias[n0 + laneM] : 0.0f;
#pragma unroll
  for (int r = 0; r < 8; ++r)
    C[(size_t)(m0 + r + hi * 8) * N + n0 + laneM] = acc[r] + bv;
}

// ---------- BiLSTM encoder recurrence: grid=2 (dir), 256 thr (8 waves) ----------
// Xf/Xb: [64*16, 2048] pre-projected (incl bias). Whh*: [2048,512] bf16. h0/c0: [2,16,512] f32.
__global__ __launch_bounds__(256) void enc_rnn_kernel(
    const float* __restrict__ Xf, const float* __restrict__ Xb,
    const u16* __restrict__ Whhf, const u16* __restrict__ Whhb,
    const float* __restrict__ h0, const float* __restrict__ c0,
    u16* __restrict__ hOut, float* __restrict__ cOut) {
  __shared__ __align__(16) u16 h_sh[16 * 512];
  int dir = blockIdx.x;
  const float* Xpre = dir ? Xb : Xf;
  const u16*  Whh   = dir ? Whhb : Whhf;
  int tid = threadIdx.x, wave = tid >> 5, lane = tid & 31;
  int laneM = lane & 15, hi = lane >> 4;

  for (int i = tid; i < 16 * 512; i += 256) h_sh[i] = f2bf(h0[dir * 8192 + i]);
  float cReg[4][8];
#pragma unroll
  for (int t = 0; t < 4; ++t) {
    int hid0 = (wave * 4 + t) * 16;
#pragma unroll
    for (int r = 0; r < 8; ++r)
      cReg[t][r] = c0[dir * 8192 + (r + hi * 8) * 512 + hid0 + laneM];
  }
  __syncthreads();

  for (int step = 0; step < 64; ++step) {
    int xrow = (dir ? (63 - step) : step) * 16;
    float hNew[4][8];
#pragma unroll
    for (int t = 0; t < 4; ++t) {
      int hid0 = (wave * 4 + t) * 16;
      v8f g[4];
#pragma unroll
      for (int gate = 0; gate < 4; ++gate) {
        int n0 = gate * 512 + hid0;
        const u16* arow = h_sh + laneM * 512 + hi * 16;
        const u16* brow = Whh + (size_t)(n0 + laneM) * 512 + hi * 16;
        v8f acc = {};
        for (int k0 = 0; k0 < 512; k0 += 32) {
          u32x4 a0 = *(const u32x4*)(arow + k0);
          u32x4 a1 = *(const u32x4*)(arow + k0 + 8);
          u32x4 b0 = *(const u32x4*)(brow + k0);
          u32x4 b1 = *(const u32x4*)(brow + k0 + 8);
          acc = wmma_bf16(a0, a1, b0, b1, acc);
        }
        const float* xp = Xpre + (size_t)(xrow + hi * 8) * 2048 + n0 + laneM;
#pragma unroll
        for (int r = 0; r < 8; ++r) acc[r] += xp[(size_t)r * 2048];
        g[gate] = acc;
      }
#pragma unroll
      for (int r = 0; r < 8; ++r) {
        float iv = sigm(g[0][r]), fv = sigm(g[1][r]);
        float gv = tanh_(g[2][r]), ov = sigm(g[3][r]);
        float cv = fv * cReg[t][r] + iv * gv;
        cReg[t][r] = cv;
        hNew[t][r] = ov * tanh_(cv);
      }
    }
    __syncthreads();   // everyone done reading old h
#pragma unroll
    for (int t = 0; t < 4; ++t) {
      int hid0 = (wave * 4 + t) * 16;
#pragma unroll
      for (int r = 0; r < 8; ++r)
        h_sh[(r + hi * 8) * 512 + hid0 + laneM] = f2bf(hNew[t][r]);
    }
    __syncthreads();   // new h visible
  }

  for (int i = tid; i < 16 * 512; i += 256) hOut[dir * 8192 + i] = h_sh[i];
#pragma unroll
  for (int t = 0; t < 4; ++t) {
    int hid0 = (wave * 4 + t) * 16;
#pragma unroll
    for (int r = 0; r < 8; ++r)
      cOut[dir * 8192 + (r + hi * 8) * 512 + hid0 + laneM] = cReg[t][r];
  }
}

// ---------- LM recurrence: grid=1, 512 thr (16 waves) ----------
// Xpre: [128*16, 4096] (incl bias). Whh: [4096,1024] bf16. hEnc bf16/cEnc f32: [2,16,512].
// ysOut: [2048, 1024] bf16.
__global__ __launch_bounds__(512) void lm_rnn_kernel(
    const float* __restrict__ Xpre, const u16* __restrict__ Whh,
    const u16* __restrict__ hEnc, const float* __restrict__ cEnc,
    u16* __restrict__ ysOut) {
  __shared__ __align__(16) u16 h_sh[16 * 1024];
  int tid = threadIdx.x, wave = tid >> 5, lane = tid & 31;
  int laneM = lane & 15, hi = lane >> 4;

  // torch-style reshape [2,16,512] -> [16,1024]: h_lm[row,col] = enc[row/8][(row%8)*2 + (col>=512)][col%512]
  for (int i = tid; i < 16 * 1024; i += 512) {
    int row = i >> 10, col = i & 1023;
    int src = (row >> 3) * 8192 + ((((row & 7) << 1) | (col >= 512 ? 1 : 0)) * 512) + (col & 511);
    h_sh[i] = hEnc[src];
  }
  float cReg[4][8];
#pragma unroll
  for (int t = 0; t < 4; ++t) {
    int hid0 = (wave * 4 + t) * 16;
    int col = hid0 + laneM;
#pragma unroll
    for (int r = 0; r < 8; ++r) {
      int row = r + hi * 8;
      int src = (row >> 3) * 8192 + ((((row & 7) << 1) | (col >= 512 ? 1 : 0)) * 512) + (col & 511);
      cReg[t][r] = cEnc[src];
    }
  }
  __syncthreads();

  for (int step = 0; step < 128; ++step) {
    int mrow = step * 16;
    float hNew[4][8];
#pragma unroll
    for (int t = 0; t < 4; ++t) {
      int hid0 = (wave * 4 + t) * 16;
      v8f g[4];
#pragma unroll
      for (int gate = 0; gate < 4; ++gate) {
        int n0 = gate * 1024 + hid0;
        const u16* arow = h_sh + laneM * 1024 + hi * 16;
        const u16* brow = Whh + (size_t)(n0 + laneM) * 1024 + hi * 16;
        v8f acc = {};
        for (int k0 = 0; k0 < 1024; k0 += 32) {
          u32x4 a0 = *(const u32x4*)(arow + k0);
          u32x4 a1 = *(const u32x4*)(arow + k0 + 8);
          u32x4 b0 = *(const u32x4*)(brow + k0);
          u32x4 b1 = *(const u32x4*)(brow + k0 + 8);
          acc = wmma_bf16(a0, a1, b0, b1, acc);
        }
        const float* xp = Xpre + (size_t)(mrow + hi * 8) * 4096 + n0 + laneM;
#pragma unroll
        for (int r = 0; r < 8; ++r) acc[r] += xp[(size_t)r * 4096];
        g[gate] = acc;
      }
#pragma unroll
      for (int r = 0; r < 8; ++r) {
        float iv = sigm(g[0][r]), fv = sigm(g[1][r]);
        float gv = tanh_(g[2][r]), ov = sigm(g[3][r]);
        float cv = fv * cReg[t][r] + iv * gv;
        cReg[t][r] = cv;
        hNew[t][r] = ov * tanh_(cv);
      }
    }
    __syncthreads();
#pragma unroll
    for (int t = 0; t < 4; ++t) {
      int hid0 = (wave * 4 + t) * 16;
#pragma unroll
      for (int r = 0; r < 8; ++r) {
        u16 hv = f2bf(hNew[t][r]);
        int row = r + hi * 8;
        h_sh[row * 1024 + hid0 + laneM] = hv;
        ysOut[(size_t)(mrow + row) * 1024 + hid0 + laneM] = hv;
      }
    }
    __syncthreads();
  }
}

// ---------- decoder: out[2048, 50257] = ys[2048,1024] @ Wdec[50257,1024]^T + bdec ----------
// One block per 128 vocab columns; Wdec slab staged once into 256KB LDS as bf16.
__global__ __launch_bounds__(256) void decode_kernel(
    const u16* __restrict__ ys, const float* __restrict__ Wdec,
    const float* __restrict__ bdec, float* __restrict__ out) {
  extern __shared__ __align__(16) u16 wTile[];   // [128][1024] bf16 = 256 KB
  const int V = 50257, K = 1024;
  int tid = threadIdx.x;
  int nBase = blockIdx.x * 128;

  for (int i = tid; i < 128 * 1024 / 4; i += 256) {
    int f = i * 4;
    int row = f >> 10;
    int col = f & 1023;
    int gRow = nBase + row;
    f32x4 v = {0.0f, 0.0f, 0.0f, 0.0f};
    if (gRow < V) v = *(const f32x4*)(Wdec + (size_t)gRow * K + col);
    u32x2 p;
    p.x = ((unsigned int)f2bf(v.y) << 16) | f2bf(v.x);
    p.y = ((unsigned int)f2bf(v.w) << 16) | f2bf(v.z);
    *(u32x2*)(wTile + row * 1024 + col) = p;
  }
  __syncthreads();

  int wave = tid >> 5, lane = tid & 31;
  int laneM = lane & 15, hi = lane >> 4;
  int colLocal = wave * 16 + laneM;
  int colG = nBase + colLocal;
  bool valid = colG < V;
  float bv = valid ? bdec[colG] : 0.0f;
  const u16* brow = wTile + colLocal * 1024 + hi * 16;

  for (int mt = 0; mt < 128; ++mt) {
    int m0 = mt * 16;
    const u16* arow = ys + (size_t)(m0 + laneM) * 1024 + hi * 16;
    v8f acc = {};
    for (int k0 = 0; k0 < 1024; k0 += 32) {
      u32x4 a0 = *(const u32x4*)(arow + k0);
      u32x4 a1 = *(const u32x4*)(arow + k0 + 8);
      u32x4 b0 = *(const u32x4*)(brow + k0);
      u32x4 b1 = *(const u32x4*)(brow + k0 + 8);
      acc = wmma_bf16(a0, a1, b0, b1, acc);
    }
    if (valid) {
#pragma unroll
      for (int r = 0; r < 8; ++r)
        out[(size_t)(m0 + r + hi * 8) * V + colG] = acc[r] + bv;
    }
  }
}

// ---------- host ----------
extern "C" void kernel_launch(void* const* d_in, const int* in_sizes, int n_in,
                              void* d_out, int out_size, void* d_ws, size_t ws_size,
                              hipStream_t stream) {
  (void)in_sizes; (void)n_in; (void)out_size; (void)ws_size;
  const int*   x        = (const int*)d_in[0];
  const int*   table    = (const int*)d_in[1];
  const float* enc_h0   = (const float*)d_in[2];
  const float* enc_c0   = (const float*)d_in[3];
  const float* embed    = (const float*)d_in[6];
  const float* t_embed  = (const float*)d_in[7];
  const float* Wih_f    = (const float*)d_in[8];
  const float* Whh_f    = (const float*)d_in[9];
  const float* b_f      = (const float*)d_in[10];
  const float* Wih_b    = (const float*)d_in[11];
  const float* Whh_b    = (const float*)d_in[12];
  const float* b_b      = (const float*)d_in[13];
  const float* Wih_lm   = (const float*)d_in[14];
  const float* Whh_lm   = (const float*)d_in[15];
  const float* b_lm     = (const float*)d_in[16];
  const float* Wdec     = (const float*)d_in[17];
  const float* bdec     = (const float*)d_in[18];
  float* out = (float*)d_out;

  char* ws = (char*)d_ws;
  size_t off = 0;
  auto alloc = [&](size_t bytes) -> void* {
    void* p = ws + off;
    off = (off + bytes + 255) & ~(size_t)255;
    return p;
  };
  u16*   temb_bf   = (u16*)alloc((size_t)1024 * 512 * 2);
  u16*   emb_bf    = (u16*)alloc((size_t)2048 * 512 * 2);
  u16*   Wihf_bf   = (u16*)alloc((size_t)2048 * 512 * 2);
  u16*   Whhf_bf   = (u16*)alloc((size_t)2048 * 512 * 2);
  u16*   Wihb_bf   = (u16*)alloc((size_t)2048 * 512 * 2);
  u16*   Whhb_bf   = (u16*)alloc((size_t)2048 * 512 * 2);
  u16*   Wihlm_bf  = (u16*)alloc((size_t)4096 * 512 * 2);
  u16*   Whhlm_bf  = (u16*)alloc((size_t)4096 * 1024 * 2);
  float* Xf        = (float*)alloc((size_t)1024 * 2048 * 4);
  float* Xb        = (float*)alloc((size_t)1024 * 2048 * 4);
  float* Xlm       = (float*)alloc((size_t)2048 * 4096 * 4);
  u16*   hEnc_bf   = (u16*)alloc((size_t)2 * 16 * 512 * 2);
  float* cEnc      = (float*)alloc((size_t)2 * 16 * 512 * 4);
  u16*   ys_bf     = (u16*)alloc((size_t)2048 * 1024 * 2);

  const int thr = 256;
  auto gi = [](long long n, int t) { return (unsigned)((n + t - 1) / t); };

  // weight conversions f32 -> bf16
  cvt_bf16_kernel<<<gi(2048LL*512, thr), thr, 0, stream>>>(Wih_f,  Wihf_bf,  2048LL*512);
  cvt_bf16_kernel<<<gi(2048LL*512, thr), thr, 0, stream>>>(Whh_f,  Whhf_bf,  2048LL*512);
  cvt_bf16_kernel<<<gi(2048LL*512, thr), thr, 0, stream>>>(Wih_b,  Wihb_bf,  2048LL*512);
  cvt_bf16_kernel<<<gi(2048LL*512, thr), thr, 0, stream>>>(Whh_b,  Whhb_bf,  2048LL*512);
  cvt_bf16_kernel<<<gi(4096LL*512, thr), thr, 0, stream>>>(Wih_lm, Wihlm_bf, 4096LL*512);
  cvt_bf16_kernel<<<gi(4096LL*1024,thr), thr, 0, stream>>>(Whh_lm, Whhlm_bf, 4096LL*1024);

  // embedding gathers (time-major rows l*B+b)
  gather_embed_kernel<<<gi(64LL*16*512,  thr), thr, 0, stream>>>(table, 16, 64,  t_embed, temb_bf, 512);
  gather_embed_kernel<<<gi(128LL*16*512, thr), thr, 0, stream>>>(x,     16, 128, embed,   emb_bf,  512);

  // input-projection precompute GEMMs (bias folded in)
  gemm_bf16_kernel<<<dim3(64, 32),  128, 0, stream>>>(temb_bf, Wihf_bf,  b_f,  Xf,  512, 2048);
  gemm_bf16_kernel<<<dim3(64, 32),  128, 0, stream>>>(temb_bf, Wihb_bf,  b_b,  Xb,  512, 2048);
  gemm_bf16_kernel<<<dim3(128, 64), 128, 0, stream>>>(emb_bf,  Wihlm_bf, b_lm, Xlm, 512, 4096);

  // recurrences
  enc_rnn_kernel<<<2, 256, 0, stream>>>(Xf, Xb, Whhf_bf, Whhb_bf, enc_h0, enc_c0, hEnc_bf, cEnc);
  lm_rnn_kernel<<<1, 512, 0, stream>>>(Xlm, Whhlm_bf, hEnc_bf, cEnc, ys_bf);

  // dominant decode GEMM with 256KB LDS-resident Wdec slab
  (void)hipFuncSetAttribute((const void*)decode_kernel,
                            hipFuncAttributeMaxDynamicSharedMemorySize, 128 * 1024 * 2);
  decode_kernel<<<393, 256, 128 * 1024 * 2, stream>>>(ys_bf, Wdec, bdec, out);
}
